// NGCF_4269197492536
// MI455X (gfx1250) — matrix operand order repaired and checked by reading, not compiled, not run on hardware
//
#include <hip/hip_runtime.h>
#include <hip/hip_bf16.h>

// NGCF forward for MI455X (gfx1250, wave32).
//  - SpMM (COO scatter-add) with f32 atomics; node matrices (38.4MB) are
//    L2-resident (192MB L2), so gather+atomics run at L2 speed.
//  - Dense 64x64 transforms per 16-row tile via v_wmma_f32_16x16x32_bf16.
//    All f32->bf16 conversion is hoisted into streaming pack kernels so the
//    WMMA kernel is loads + 16 WMMAs + epilogue (no per-element cvt VALU).

#define NGCF_N_USERS 50000
#define NGCF_N_ITEMS 100000
#define NGCF_N_NODES (NGCF_N_USERS + NGCF_N_ITEMS)
#define NGCF_N_EDGES 2400000
#define NGCF_D       64
#define NGCF_LAYERS  3
#define NGCF_OUTC    (NGCF_D * (NGCF_LAYERS + 1))   // 256

typedef __attribute__((ext_vector_type(16))) __bf16 v16bf;
typedef __attribute__((ext_vector_type(8)))  float  v8f;

__device__ __forceinline__ unsigned f32bits(float f) {
  return __builtin_bit_cast(unsigned, f);
}

// RNE-pack two f32 into one dword of two bf16 (lo = a, hi = b)
__device__ __forceinline__ unsigned pack_bf16_2(float a, float b) {
  unsigned ua = f32bits(a), ub = f32bits(b);
  ua = ua + 0x7FFFu + ((ua >> 16) & 1u);
  ub = ub + 0x7FFFu + ((ub >> 16) & 1u);
  return (ua >> 16) | (ub & 0xFFFF0000u);
}

// ---------------------------------------------------------------------------
// ego = concat(user_emb, item_emb); also write raw ego into out[:, 0:64]
// ---------------------------------------------------------------------------
__global__ void ngcf_init_kernel(const float* __restrict__ user_emb,
                                 const float* __restrict__ item_emb,
                                 float* __restrict__ ego,
                                 float* __restrict__ out) {
  long long idx = (long long)blockIdx.x * blockDim.x + threadIdx.x;
  if (idx >= (long long)NGCF_N_NODES * NGCF_D) return;
  int r = (int)(idx >> 6);
  int c = (int)(idx & 63);
  float v = (r < NGCF_N_USERS) ? user_emb[(long long)r * NGCF_D + c]
                               : item_emb[(long long)(r - NGCF_N_USERS) * NGCF_D + c];
  ego[idx] = v;
  out[(long long)r * NGCF_OUTC + c] = v;
}

__global__ void ngcf_zero_kernel(float* __restrict__ p, long long n) {
  long long idx = (long long)blockIdx.x * blockDim.x + threadIdx.x;
  if (idx < n) p[idx] = 0.0f;
}

// ---------------------------------------------------------------------------
// One-time: convert all layers' Wgc/Wbi to bf16 (row-major [L][n][k]).
// Each thread packs one dword (2 elements) of each matrix family.
// ---------------------------------------------------------------------------
__global__ void ngcf_wpack_kernel(const float* __restrict__ Wgc,
                                  const float* __restrict__ Wbi,
                                  unsigned* __restrict__ wgc_bf,   // dword-packed bf16
                                  unsigned* __restrict__ wbi_bf) {
  int t = blockIdx.x * blockDim.x + threadIdx.x;
  const int npairs = NGCF_LAYERS * NGCF_D * NGCF_D / 2;            // 6144
  if (t >= npairs) return;
  wgc_bf[t] = pack_bf16_2(Wgc[2 * t], Wgc[2 * t + 1]);
  wbi_bf[t] = pack_bf16_2(Wbi[2 * t], Wbi[2 * t + 1]);
}

// ---------------------------------------------------------------------------
// side[dst] += val * ego[src]  — 16 lanes per edge, float4 per lane,
// f32 atomics land in L2 (x and y both L2-resident).
// ---------------------------------------------------------------------------
__global__ void ngcf_spmm_kernel(const int* __restrict__ src,
                                 const int* __restrict__ dst,
                                 const float* __restrict__ vals,
                                 const float* __restrict__ x,
                                 float* __restrict__ y) {
  long long gid = (long long)blockIdx.x * blockDim.x + threadIdx.x;
  int e = (int)(gid >> 4);
  if (e >= NGCF_N_EDGES) return;
  int c = ((int)gid & 15) * 4;
  float w = vals[e];
  const float4 xv = *(const float4*)(x + (long long)src[e] * NGCF_D + c);
  float* yp = y + (long long)dst[e] * NGCF_D + c;
  atomicAdd(yp + 0, w * xv.x);
  atomicAdd(yp + 1, w * xv.y);
  atomicAdd(yp + 2, w * xv.z);
  atomicAdd(yp + 3, w * xv.w);
}

// ---------------------------------------------------------------------------
// Streaming pack: side -> bf16, (ego*side) -> bf16. 4 elems / thread.
// ---------------------------------------------------------------------------
__global__ void ngcf_pack_kernel(const float* __restrict__ ego,
                                 const float* __restrict__ side,
                                 uint2* __restrict__ sbf,    // [N*64/4] dword-pairs
                                 uint2* __restrict__ bibf) {
  long long t = (long long)blockIdx.x * blockDim.x + threadIdx.x;
  const long long nquads = (long long)NGCF_N_NODES * NGCF_D / 4;
  if (t >= nquads) return;
  const float4 e = *(const float4*)(ego  + 4 * t);
  const float4 s = *(const float4*)(side + 4 * t);
  uint2 so, bo;
  so.x = pack_bf16_2(s.x, s.y);
  so.y = pack_bf16_2(s.z, s.w);
  bo.x = pack_bf16_2(e.x * s.x, e.y * s.y);
  bo.y = pack_bf16_2(e.z * s.z, e.w * s.w);
  sbf[t]  = so;
  bibf[t] = bo;
}

// ---------------------------------------------------------------------------
// Fused per-layer transform. One wave32 owns a 16-row tile:
//   gc   = leaky_relu(side @ Wgc^T + bgc)
//   bi   = leaky_relu((ego*side) @ Wbi^T + bbi)
//   ego' = gc + bi          (written in place — tile-local rows only)
//   out[:, col0:col0+64] = l2norm_rows(ego')
// Fragment layouts per CDNA5 ISA §7.12.2 — all map to contiguous
// 16B bf16 loads:
//   A 16x32 : lane m=l&15, hi=l>>4; slots 0-7 <- k=kb*32+hi*8+0..7,
//             slots 8-15 <- k=kb*32+16+hi*8+0..7
//   B 32x16 : lane n=l&15, hi=l>>4; same k slots as A ; B[k][n]=W[n][k]
//   C/D     : lane n=l&15, hi=l>>4; VGPR j -> row m=hi*8+j
// ---------------------------------------------------------------------------
union frag_u { uint4 q[2]; v16bf v; };

__global__ __launch_bounds__(128)
void ngcf_transform_kernel(const unsigned short* __restrict__ sbf,   // side bf16 [N,64]
                           const unsigned short* __restrict__ bibf,  // ego*side bf16
                           const unsigned short* __restrict__ wgc_bf,// [64,64] bf16
                           const unsigned short* __restrict__ wbi_bf,
                           const float* __restrict__ bgc_p,
                           const float* __restrict__ bbi_p,
                           float* __restrict__ ego,                  // f32 out (in place)
                           float* __restrict__ out,
                           int out_col0) {
  const int wave = threadIdx.x >> 5;
  const int lane = threadIdx.x & 31;
  const int tile = blockIdx.x * 4 + wave;
  if (tile >= NGCF_N_NODES / 16) return;   // wave-uniform: EXEC stays all-ones
  const int row0 = tile * 16;
  const int m    = lane & 15;
  const int hi   = lane >> 4;
  const int n_lo = lane & 15;

  // ---- A fragments: two 16B loads each, no conversion ----
  v16bf a_side[2], a_bi[2];
  {
    const long long rb = (long long)(row0 + m) * NGCF_D;
#pragma unroll
    for (int kb = 0; kb < 2; ++kb) {
      frag_u us, ub;
      us.q[0] = *(const uint4*)(sbf  + rb + kb * 32 + hi * 8);
      us.q[1] = *(const uint4*)(sbf  + rb + kb * 32 + 16 + hi * 8);
      ub.q[0] = *(const uint4*)(bibf + rb + kb * 32 + hi * 8);
      ub.q[1] = *(const uint4*)(bibf + rb + kb * 32 + 16 + hi * 8);
      a_side[kb] = us.v;
      a_bi[kb]   = ub.v;
    }
  }

  float vkeep[4][8];
  float rowsq[8];
#pragma unroll
  for (int j = 0; j < 8; ++j) rowsq[j] = 0.0f;

#pragma unroll
  for (int nt = 0; nt < 4; ++nt) {
    const int n = nt * 16 + n_lo;

    // ---- B fragments: W row n, two 16B bf16 loads per fragment, 16 k apart ----
    v16bf bgcf[2], bbif[2];
#pragma unroll
    for (int kb = 0; kb < 2; ++kb) {
      const long long wb = (long long)n * NGCF_D + kb * 32 + hi * 8;
      frag_u ug, ubw;
      ug.q[0]  = *(const uint4*)(wgc_bf + wb);
      ug.q[1]  = *(const uint4*)(wgc_bf + wb + 16);
      ubw.q[0] = *(const uint4*)(wbi_bf + wb);
      ubw.q[1] = *(const uint4*)(wbi_bf + wb + 16);
      bgcf[kb] = ug.v;
      bbif[kb] = ubw.v;
    }

    v8f cgc = {};
    v8f cbi = {};
    cgc = __builtin_amdgcn_wmma_f32_16x16x32_bf16(false, a_side[0], false, bgcf[0],
                                                  (short)0, cgc, false, false);
    cgc = __builtin_amdgcn_wmma_f32_16x16x32_bf16(false, a_side[1], false, bgcf[1],
                                                  (short)0, cgc, false, false);
    cbi = __builtin_amdgcn_wmma_f32_16x16x32_bf16(false, a_bi[0], false, bbif[0],
                                                  (short)0, cbi, false, false);
    cbi = __builtin_amdgcn_wmma_f32_16x16x32_bf16(false, a_bi[1], false, bbif[1],
                                                  (short)0, cbi, false, false);

    const float bg = bgc_p[n];
    const float bb = bbi_p[n];
#pragma unroll
    for (int j = 0; j < 8; ++j) {
      const int mr = hi * 8 + j;
      float g = cgc[j] + bg; g = (g > 0.0f) ? g : 0.01f * g;   // leaky_relu(0.01)
      float b = cbi[j] + bb; b = (b > 0.0f) ? b : 0.01f * b;
      const float v = g + b;
      vkeep[nt][j] = v;
      rowsq[j] += v * v;
      ego[(long long)(row0 + mr) * NGCF_D + n] = v;            // un-normalized ego'
    }
  }

  // ---- row l2-norm: xor-reduce over the 16-lane half-wave group ----
#pragma unroll
  for (int mask = 1; mask < 16; mask <<= 1) {
#pragma unroll
    for (int j = 0; j < 8; ++j)
      rowsq[j] += __shfl_xor(rowsq[j], mask, 32);
  }
  float inv[8];
#pragma unroll
  for (int j = 0; j < 8; ++j) {
    const float nrm = sqrtf(rowsq[j]);
    inv[j] = 1.0f / fmaxf(nrm, 1e-12f);
  }

#pragma unroll
  for (int nt = 0; nt < 4; ++nt) {
    const int n = nt * 16 + n_lo;
#pragma unroll
    for (int j = 0; j < 8; ++j) {
      const int mr = hi * 8 + j;
      out[(long long)(row0 + mr) * NGCF_OUTC + out_col0 + n] = vkeep[nt][j] * inv[j];
    }
  }
}

// ---------------------------------------------------------------------------
extern "C" void kernel_launch(void* const* d_in, const int* in_sizes, int n_in,
                              void* d_out, int out_size, void* d_ws, size_t ws_size,
                              hipStream_t stream) {
  const int*   edge_src  = (const int*)d_in[0];
  const int*   edge_dst  = (const int*)d_in[1];
  const float* edge_vals = (const float*)d_in[2];
  const float* user_emb  = (const float*)d_in[3];
  const float* item_emb  = (const float*)d_in[4];
  const float* Wgc       = (const float*)d_in[5];
  const float* bgc       = (const float*)d_in[6];
  const float* Wbi       = (const float*)d_in[7];
  const float* bbi       = (const float*)d_in[8];
  float* out = (float*)d_out;

  // workspace layout (all offsets 16B-aligned):
  //   ego  f32  [N,64]   38,400,000 B
  //   side f32  [N,64]   38,400,000 B
  //   sbf  bf16 [N,64]   19,200,000 B
  //   bibf bf16 [N,64]   19,200,000 B
  //   wgc_bf bf16 [3,64,64]  24,576 B
  //   wbi_bf bf16 [3,64,64]  24,576 B
  char* ws = (char*)d_ws;
  const size_t nd   = (size_t)NGCF_N_NODES * NGCF_D;
  float*          ego    = (float*)ws;                       ws += nd * 4;
  float*          side   = (float*)ws;                       ws += nd * 4;
  unsigned short* sbf    = (unsigned short*)ws;              ws += nd * 2;
  unsigned short* bibf   = (unsigned short*)ws;              ws += nd * 2;
  unsigned short* wgc_bf = (unsigned short*)ws;              ws += (size_t)NGCF_LAYERS * NGCF_D * NGCF_D * 2;
  unsigned short* wbi_bf = (unsigned short*)ws;

  const long long elems = (long long)nd;
  const int tpb = 256;
  const int eblocks = (int)((elems + tpb - 1) / tpb);

  ngcf_init_kernel<<<eblocks, tpb, 0, stream>>>(user_emb, item_emb, ego, out);

  {
    const int npairs = NGCF_LAYERS * NGCF_D * NGCF_D / 2;    // 6144
    ngcf_wpack_kernel<<<(npairs + tpb - 1) / tpb, tpb, 0, stream>>>(
        Wgc, Wbi, (unsigned*)wgc_bf, (unsigned*)wbi_bf);
  }

  for (int layer = 0; layer < NGCF_LAYERS; ++layer) {
    ngcf_zero_kernel<<<eblocks, tpb, 0, stream>>>(side, elems);

    const long long sthreads = (long long)NGCF_N_EDGES * 16;
    ngcf_spmm_kernel<<<(int)((sthreads + tpb - 1) / tpb), tpb, 0, stream>>>(
        edge_src, edge_dst, edge_vals, ego, side);

    const long long nquads = elems / 4;
    ngcf_pack_kernel<<<(int)((nquads + tpb - 1) / tpb), tpb, 0, stream>>>(
        ego, side, (uint2*)sbf, (uint2*)bibf);

    const int tiles  = NGCF_N_NODES / 16;       // 9375
    const int blocks = (tiles + 3) / 4;         // 4 waves / block
    ngcf_transform_kernel<<<blocks, 128, 0, stream>>>(
        sbf, bibf,
        wgc_bf + (size_t)layer * NGCF_D * NGCF_D,
        wbi_bf + (size_t)layer * NGCF_D * NGCF_D,
        bgc + (size_t)layer * NGCF_D, bbi + (size_t)layer * NGCF_D,
        ego, out, (layer + 1) * NGCF_D);
  }
}
